// TapGRU_28381143892185
// MI455X (gfx1250) — compile-verified, matrix operand order checked by reading, not run on hardware
//
#include <hip/hip_runtime.h>
#include <hip/hip_bf16.h>
#include <math.h>

// Problem dims (fixed by the reference)
#define T_DIM 512
#define B_DIM 64
#define I_DIM 512
#define D_DIM 1024

// Persistent recurrence kernel grid size (must be co-resident; 32 blocks of
// 256 threads is trivially resident on any MI-class part)
#define NBLK_STEP 32

typedef __attribute__((ext_vector_type(16))) __bf16 v16bf;
typedef __attribute__((ext_vector_type(8)))  __bf16 v8bf;
typedef __attribute__((ext_vector_type(8)))  float  v8f;

__device__ __forceinline__ unsigned short f32_to_bf16_rne(float f) {
  unsigned int u = __builtin_bit_cast(unsigned int, f);
  u += 0x7FFFu + ((u >> 16) & 1u);   // round to nearest even
  return (unsigned short)(u >> 16);
}
__device__ __forceinline__ float bf16_to_f32(unsigned short h) {
  return __builtin_bit_cast(float, (unsigned int)h << 16);
}

// Load one 16x32 bf16 A/B fragment for v_wmma_f32_16x16x32_bf16.
// CDNA5 layout: lanes 0-15 hold row (lane&15), K = k..k+7 and k+16..k+23;
// lanes 16-31 hold K = k+8..15 and k+24..31 (phase folded into caller's k).
// Two 16-byte global loads per lane.
__device__ __forceinline__ v16bf load_frag16(const unsigned short* __restrict__ base,
                                             int row, int ld, int k) {
  const __bf16* p = reinterpret_cast<const __bf16*>(base + (long)row * ld + k);
  v8bf lo = *reinterpret_cast<const v8bf*>(p);        // K k..k+7
  v8bf hi = *reinterpret_cast<const v8bf*>(p + 16);   // K k+16..k+23
  return __builtin_shufflevector(lo, hi, 0,1,2,3,4,5,6,7,8,9,10,11,12,13,14,15);
}

__device__ __forceinline__ v8f wmma_bf16(v16bf a, v16bf b, v8f c) {
  return __builtin_amdgcn_wmma_f32_16x16x32_bf16(false, a, false, b, (short)0, c,
                                                 false, false);
}

// ---------------------------------------------------------------- converts --

__global__ void k_f32_to_bf16(const float* __restrict__ src,
                              unsigned short* __restrict__ dst, int n) {
  int i = blockIdx.x * blockDim.x + threadIdx.x;
  if (i < n) dst[i] = f32_to_bf16_rne(src[i]);
}

// src [K,N] f32 row-major -> dst [N,K] bf16 (WMMA B operand contiguous in K).
__global__ void k_transpose_bf16(const float* __restrict__ src,
                                 unsigned short* __restrict__ dst, int K, int N) {
  long i = (long)blockIdx.x * blockDim.x + threadIdx.x;
  long total = (long)K * N;
  if (i < total) {
    int n = (int)(i % N);
    int k = (int)(i / N);
    dst[(long)n * K + k] = f32_to_bf16_rne(src[i]);
  }
}

__global__ void k_zero_bf16(unsigned short* __restrict__ dst, int n) {
  int i = blockIdx.x * blockDim.x + threadIdx.x;
  if (i < n) dst[i] = 0;
}

__global__ void k_zero_u32(unsigned int* __restrict__ dst, int n) {
  int i = blockIdx.x * blockDim.x + threadIdx.x;
  if (i < n) dst[i] = 0u;
}

// --------------------------------------------------------------- big GEMM --
// C[M,N] = A[M,K](bf16) @ Bt[N,K](bf16)^T + bias[N]
// Block: 256 threads = 8 waves; block tile 128(M) x 64(N).
// Each wave: one 16-row stripe x 64 cols = 4 accumulators, A fragment reused 4x.
__global__ void k_gemm_bf16(const unsigned short* __restrict__ A,
                            const unsigned short* __restrict__ Bt,
                            const float* __restrict__ bias,
                            float* __restrict__ C,
                            int M, int N, int K) {
  const int lane = threadIdx.x & 31;
  const int wave = threadIdx.x >> 5;
  const int m0   = blockIdx.y * 128 + wave * 16;
  const int n0   = blockIdx.x * 64;
  const int rA   = lane & 15;          // fragment row / col within tile
  const int kh   = (lane >> 4) * 8;    // K phase for upper half-wave

  v8f acc[4] = {};
  for (int k0 = 0; k0 < K; k0 += 32) {
    v16bf a = load_frag16(A, m0 + rA, K, k0 + kh);
#pragma unroll
    for (int j = 0; j < 4; ++j) {
      v16bf b = load_frag16(Bt, n0 + j * 16 + rA, K, k0 + kh);
      acc[j] = wmma_bf16(a, b, acc[j]);
    }
  }

  // C/D layout: VGPR e -> row m0+e (lanes 0-15) / m0+8+e (lanes 16-31)
  const int mrow = m0 + (lane >> 4) * 8;
#pragma unroll
  for (int j = 0; j < 4; ++j) {
    const int col = n0 + j * 16 + (lane & 15);
    const float bv = bias ? bias[col] : 0.0f;
#pragma unroll
    for (int e = 0; e < 8; ++e)
      C[(long)(mrow + e) * N + col] = acc[j][e] + bv;
  }
}

// ----------------------------------------------- persistent GRU recurrence --
// Single kernel for all T=512 steps. Grid: 32 blocks x 256 threads (8 waves).
// Wave (mtile 0..3, ntile 0..1) owns the 16x16 tile (m0, n0) of h_new and
// accumulates THREE GEMM tiles over K=1024: r-preact (U cols n), u-preact
// (U cols D+n), candidate (Ux cols n). U/Ux bf16 (6 MB total -> L2 resident).
// Steps are separated by a grid-wide release/acquire barrier on an atomic
// counter (monotonic: target = (t+1)*NBLK_STEP), with double-buffered bf16 h.
__global__ void k_gru_persistent(const unsigned short* __restrict__ Ut,   // [2D,D] bf16
                                 const unsigned short* __restrict__ Uxt,  // [D,D]  bf16
                                 const float* __restrict__ x_all,         // [T,B,2D]
                                 const float* __restrict__ xx_all,        // [T,B,D]
                                 const float* __restrict__ mask,          // [T,B]
                                 unsigned short* __restrict__ h0,         // [B,D] bf16
                                 unsigned short* __restrict__ h1,         // [B,D] bf16
                                 float* __restrict__ out,                 // [T,B,D]
                                 unsigned int* __restrict__ counter) {
  const int lane  = threadIdx.x & 31;
  const int wave  = threadIdx.x >> 5;
  const int m0    = (wave >> 1) * 16;                  // batch tile (B=64 -> 4 tiles)
  const int n0    = blockIdx.x * 32 + (wave & 1) * 16; // hidden-dim tile
  const int rA    = lane & 15;
  const int kh    = (lane >> 4) * 8;
  const int mbase = m0 + (lane >> 4) * 8;
  const int col   = n0 + (lane & 15);

  for (int t = 0; t < T_DIM; ++t) {
    const unsigned short* h_in  = (t & 1) ? h1 : h0;
    unsigned short*       h_out = (t & 1) ? h0 : h1;
    const float* x_t    = x_all  + (size_t)t * B_DIM * 2 * D_DIM;
    const float* xx_t   = xx_all + (size_t)t * B_DIM * D_DIM;
    const float* mask_t = mask   + (size_t)t * B_DIM;
    float*       out_t  = out    + (size_t)t * B_DIM * D_DIM;

    v8f accR = {}, accU = {}, accC = {};
    for (int k0 = 0; k0 < D_DIM; k0 += 32) {
      v16bf a  = load_frag16(h_in, m0 + rA, D_DIM, k0 + kh);
      v16bf bR = load_frag16(Ut,  n0 + rA,         D_DIM, k0 + kh);
      v16bf bU = load_frag16(Ut,  D_DIM + n0 + rA, D_DIM, k0 + kh);
      v16bf bC = load_frag16(Uxt, n0 + rA,         D_DIM, k0 + kh);
      accR = wmma_bf16(a, bR, accR);
      accU = wmma_bf16(a, bU, accU);
      accC = wmma_bf16(a, bC, accC);
    }

#pragma unroll
    for (int e = 0; e < 8; ++e) {
      const int m = mbase + e;
      const float xr = x_t[(long)m * (2 * D_DIM) + col];
      const float xu = x_t[(long)m * (2 * D_DIM) + D_DIM + col];
      const float xx = xx_t[(long)m * D_DIM + col];
      const float r  = 1.0f / (1.0f + __expf(-(accR[e] + xr)));
      const float u  = 1.0f / (1.0f + __expf(-(accU[e] + xu)));
      const float hp = bf16_to_f32(h_in[(long)m * D_DIM + col]);
      const float z  = accC[e] * r + xx;
      const float e2 = __expf(2.0f * z);
      const float hc = (e2 - 1.0f) / (e2 + 1.0f);       // tanh
      float hn = u * hp + (1.0f - u) * hc;
      const float mv = mask_t[m];
      hn = mv * hn + (1.0f - mv) * hp;
      h_out[(long)m * D_DIM + col] = f32_to_bf16_rne(hn);
      out_t[(long)m * D_DIM + col] = hn;
    }

    // ---- grid-wide barrier: release h_out, then acquire all blocks' writes
    __threadfence();          // make h_out visible device-wide (release)
    __syncthreads();
    if (threadIdx.x == 0) {
      atomicAdd(counter, 1u);
      const unsigned int target = (unsigned int)(t + 1) * NBLK_STEP;
      while (__hip_atomic_load(counter, __ATOMIC_RELAXED,
                               __HIP_MEMORY_SCOPE_AGENT) < target) {
        __builtin_amdgcn_s_sleep(1);
      }
    }
    __syncthreads();
    __threadfence();          // acquire side
  }
}

// ------------------------------------------------------------------- launch --

extern "C" void kernel_launch(void* const* d_in, const int* in_sizes, int n_in,
                              void* d_out, int out_size, void* d_ws, size_t ws_size,
                              hipStream_t stream) {
  const float* sb   = (const float*)d_in[0];  // [T,B,I]
  const float* mask = (const float*)d_in[1];  // [T,B]
  const float* W    = (const float*)d_in[2];  // [I,2D]
  const float* bW   = (const float*)d_in[3];  // [2D]
  const float* Wx   = (const float*)d_in[4];  // [I,D]
  const float* bWx  = (const float*)d_in[5];  // [D]
  const float* U    = (const float*)d_in[6];  // [D,2D]
  const float* Ux   = (const float*)d_in[7];  // [D,D]
  float* out = (float*)d_out;                 // [T,B,D]

  char* ws = (char*)d_ws;
  size_t off = 0;
  auto carve = [&](size_t bytes) -> void* {
    void* p = ws + off;
    off += (bytes + 255) & ~(size_t)255;
    return p;
  };

  unsigned short* sb_bf = (unsigned short*)carve((size_t)T_DIM * B_DIM * I_DIM * 2);
  unsigned short* Wt    = (unsigned short*)carve((size_t)I_DIM * 2 * D_DIM * 2);
  unsigned short* Wxt   = (unsigned short*)carve((size_t)I_DIM * D_DIM * 2);
  unsigned short* Ut    = (unsigned short*)carve((size_t)D_DIM * 2 * D_DIM * 2);
  unsigned short* Uxt   = (unsigned short*)carve((size_t)D_DIM * D_DIM * 2);
  float* x_all  = (float*)carve((size_t)T_DIM * B_DIM * 2 * D_DIM * 4);
  float* xx_all = (float*)carve((size_t)T_DIM * B_DIM * D_DIM * 4);
  unsigned short* hbuf0 = (unsigned short*)carve((size_t)B_DIM * D_DIM * 2);
  unsigned short* hbuf1 = (unsigned short*)carve((size_t)B_DIM * D_DIM * 2);
  unsigned int*   ctr   = (unsigned int*)carve(256);

  // 0) reset barrier counter + h0 = 0 (must happen every call: graph replays)
  k_zero_u32<<<1, 64, 0, stream>>>(ctr, 1);
  {
    int n = B_DIM * D_DIM;
    k_zero_bf16<<<(n + 255) / 256, 256, 0, stream>>>(hbuf0, n);
  }
  // 1) convert state_below to bf16
  {
    int n = T_DIM * B_DIM * I_DIM;
    k_f32_to_bf16<<<(n + 255) / 256, 256, 0, stream>>>(sb, sb_bf, n);
  }
  // 2) transpose-convert weights to [N,K] bf16
  {
    int n = I_DIM * 2 * D_DIM;
    k_transpose_bf16<<<(n + 255) / 256, 256, 0, stream>>>(W, Wt, I_DIM, 2 * D_DIM);
    n = I_DIM * D_DIM;
    k_transpose_bf16<<<(n + 255) / 256, 256, 0, stream>>>(Wx, Wxt, I_DIM, D_DIM);
    n = D_DIM * 2 * D_DIM;
    k_transpose_bf16<<<(n + 255) / 256, 256, 0, stream>>>(U, Ut, D_DIM, 2 * D_DIM);
    n = D_DIM * D_DIM;
    k_transpose_bf16<<<(n + 255) / 256, 256, 0, stream>>>(Ux, Uxt, D_DIM, D_DIM);
  }
  // 3) big input-projection GEMMs (HBM-bound; bf16 WMMA)
  {
    const int M = T_DIM * B_DIM;             // 32768
    dim3 grid1(2 * D_DIM / 64, M / 128);     // (32, 256)
    k_gemm_bf16<<<grid1, 256, 0, stream>>>(sb_bf, Wt, bW, x_all, M, 2 * D_DIM, I_DIM);
    dim3 grid2(D_DIM / 64, M / 128);         // (16, 256)
    k_gemm_bf16<<<grid2, 256, 0, stream>>>(sb_bf, Wxt, bWx, xx_all, M, D_DIM, I_DIM);
  }
  // 4) whole recurrence in ONE persistent kernel (grid barrier between steps)
  k_gru_persistent<<<NBLK_STEP, 256, 0, stream>>>(Ut, Uxt, x_all, xx_all, mask,
                                                  hbuf0, hbuf1, out, ctr);

  (void)in_sizes; (void)n_in; (void)out_size; (void)ws_size;
}
